// DeepSpeedMLP_77189152243985
// MI455X (gfx1250) — compile-verified, matrix operand order checked by reading, not run on hardware
//
#include <hip/hip_runtime.h>
#include <hip/hip_bf16.h>
#include <math.h>

#define HIDDEN 4096
#define INTER  16384
#define MTOT   8192      // B*S = 4*2048 tokens
#define EPS    1e-5f

typedef __attribute__((ext_vector_type(16))) __bf16 v16bf;
typedef __attribute__((ext_vector_type(8)))  __bf16 v8bf;
typedef __attribute__((ext_vector_type(4)))  __bf16 v4bf;
typedef __attribute__((ext_vector_type(8)))  float  v8f;
typedef __attribute__((ext_vector_type(4)))  float  v4f;

// ---------------------------------------------------------------------------
// CDNA5 async global->LDS copy (ASYNCcnt-tracked). 16B per lane.
// Low 32 bits of a generic LDS pointer are the raw LDS byte address.
// ---------------------------------------------------------------------------
__device__ __forceinline__ void async_cp_b128(const void* gsrc, void* ldst) {
    unsigned lds_addr = (unsigned)(size_t)ldst;
    unsigned long long gaddr = (unsigned long long)(size_t)gsrc;
    asm volatile("global_load_async_to_lds_b128 %0, %1, off"
                 :: "v"(lds_addr), "v"(gaddr) : "memory");
}
__device__ __forceinline__ void async_wait_all() {
    asm volatile("s_wait_asynccnt 0x0" ::: "memory");
}

// ---------------------------------------------------------------------------
// Kernel 1: res = input + residual + bias ; ln = LayerNorm(res)*gamma+beta
// One block per token row; res kept fp32 (needed in final add), ln in bf16.
// ---------------------------------------------------------------------------
__global__ __launch_bounds__(256) void add_ln_kernel(
    const float* __restrict__ inp, const float* __restrict__ res,
    const float* __restrict__ bias, const float* __restrict__ gamma,
    const float* __restrict__ beta,
    float* __restrict__ res_out, __bf16* __restrict__ ln_out)
{
    const int row = blockIdx.x;
    const int tid = threadIdx.x;
    const float* ip = inp + (size_t)row * HIDDEN;
    const float* rp = res + (size_t)row * HIDDEN;

    float vals[16];
    float s = 0.f, ss = 0.f;
#pragma unroll
    for (int j = 0; j < 4; ++j) {
        const int c = (j * 256 + tid) * 4;          // coalesced 16B chunks
        v4f a  = *(const v4f*)(ip + c);
        v4f b  = *(const v4f*)(rp + c);
        v4f bi = *(const v4f*)(bias + c);
        v4f r  = a + b + bi;
        *(v4f*)&vals[j * 4] = r;
        s  += r.x + r.y + r.z + r.w;
        ss += r.x * r.x + r.y * r.y + r.z * r.z + r.w * r.w;
    }
    // wave32 reduction, then cross-wave via LDS
#pragma unroll
    for (int o = 16; o > 0; o >>= 1) {
        s  += __shfl_xor(s,  o, 32);
        ss += __shfl_xor(ss, o, 32);
    }
    __shared__ float ps[8], pss[8];
    const int wave = tid >> 5, lane = tid & 31;
    if (lane == 0) { ps[wave] = s; pss[wave] = ss; }
    __syncthreads();
    if (tid == 0) {
        float t = 0.f, tt = 0.f;
        for (int i = 0; i < 8; ++i) { t += ps[i]; tt += pss[i]; }
        float mu  = t * (1.0f / HIDDEN);
        float var = tt * (1.0f / HIDDEN) - mu * mu;
        ps[0]  = mu;
        pss[0] = rsqrtf(var + EPS);
    }
    __syncthreads();
    const float mu = ps[0], rstd = pss[0];
#pragma unroll
    for (int j = 0; j < 4; ++j) {
        const int c = (j * 256 + tid) * 4;
        v4f g  = *(const v4f*)(gamma + c);
        v4f be = *(const v4f*)(beta + c);
        v4f r  = *(v4f*)&vals[j * 4];
        *(v4f*)(res_out + (size_t)row * HIDDEN + c) = r;
        v4f y = (r - mu) * rstd * g + be;
        v4bf o;
        o[0] = (__bf16)y.x; o[1] = (__bf16)y.y; o[2] = (__bf16)y.z; o[3] = (__bf16)y.w;
        *(v4bf*)(ln_out + (size_t)row * HIDDEN + c) = o;
    }
}

// ---------------------------------------------------------------------------
// Kernel 2: fp32 [R][C] -> bf16 [C][R]  (weights to WMMA-friendly [N][K],
// and halves the HBM/L2 footprint of the streamed weights)
// ---------------------------------------------------------------------------
__global__ __launch_bounds__(256) void transpose_bf16_kernel(
    const float* __restrict__ src, __bf16* __restrict__ dst, int R, int C)
{
    __shared__ float t[32][33];
    const int tx = threadIdx.x, ty = threadIdx.y;       // 32 x 8
    const int c0 = blockIdx.x * 32, r0 = blockIdx.y * 32;
#pragma unroll
    for (int j = 0; j < 4; ++j) {
        const int r = r0 + ty + j * 8;
        t[ty + j * 8][tx] = src[(size_t)r * C + c0 + tx];
    }
    __syncthreads();
#pragma unroll
    for (int j = 0; j < 4; ++j) {
        const int c = c0 + ty + j * 8;
        dst[(size_t)c * R + r0 + tx] = (__bf16)t[tx][ty + j * 8];
    }
}

// ---------------------------------------------------------------------------
// WMMA GEMM: C[M][N] = A[M][K](bf16) * Bt[N][K](bf16)^T, f32 accumulate.
// 128x128 block tile, BK=32 (one v_wmma_f32_16x16x32_bf16 K-step).
// 8 waves (wave32): 2x4 wave grid, each wave owns 64x32 = 4x2 16x16 frags.
// Double-buffered LDS filled by async global->LDS (ASYNCcnt).
// EPI==0: x += bias[n]; gelu_tanh(x) -> bf16     (GEMM1)
// EPI==1: x += bias[n] + resid[m][n] -> fp32     (GEMM2)
// ---------------------------------------------------------------------------
template <int EPI>
__global__ __launch_bounds__(256) void gemm_wmma_kernel(
    const __bf16* __restrict__ A, const __bf16* __restrict__ Bt,
    const float* __restrict__ bias, const float* __restrict__ resid,
    float* __restrict__ Cf, __bf16* __restrict__ Cb,
    int M, int N, int K)
{
    constexpr int BM = 128, BN = 128, BK = 32;
    constexpr int LDT = 80;                 // bytes/LDS row: 64B data + 16B pad
    __shared__ char smem[2 * (BM + BN) * LDT];   // 40 KB (of 320 KB/WGP)

    const int tid   = threadIdx.x;
    const int lane  = tid & 31, wave = tid >> 5;
    const int waveM = wave & 1, waveN = wave >> 1;     // 2 x 4 wave grid
    const int bm = blockIdx.y * BM, bn = blockIdx.x * BN;

    const int mrow  = lane & 15;            // row within 16x16 frag
    const int khalf = (lane >> 4) * 8;      // lanes 0-15: K 0-7/16-23, 16-31: 8-15/24-31

    auto issue = [&](int kt, int b) {
        const char* Ag = (const char*)(A + (size_t)bm * K + (size_t)kt * BK);
        const char* Bg = (const char*)(Bt + (size_t)bn * K + (size_t)kt * BK);
        char* As = smem + b * (BM + BN) * LDT;
        char* Bs = As + BM * LDT;
#pragma unroll
        for (int j = 0; j < 2; ++j) {       // 512 16B chunks / 256 threads
            const int c = tid + j * 256;
            const int r = c >> 2, kc = (c & 3) * 8;    // kc in bf16 elems
            async_cp_b128(Ag + ((size_t)r * K + kc) * 2, As + r * LDT + kc * 2);
            async_cp_b128(Bg + ((size_t)r * K + kc) * 2, Bs + r * LDT + kc * 2);
        }
    };

    v8f acc[4][2] = {};
    const int KT = K / BK;

    issue(0, 0);
    async_wait_all();
    __syncthreads();

    for (int kt = 0; kt < KT; ++kt) {
        const int b = kt & 1;
        if (kt + 1 < KT) issue(kt + 1, b ^ 1);   // prefetch behind WMMA

        const char* As = smem + b * (BM + BN) * LDT;
        const char* Bs = As + BM * LDT;

        v16bf afrag[4], bfrag[2];
#pragma unroll
        for (int i = 0; i < 4; ++i) {       // A: 16x32 frags, two ds_load_b128 each
            const char* p = As + (waveM * 64 + i * 16 + mrow) * LDT;
            union { v16bf v; v8bf h[2]; } u;
            u.h[0] = *(const v8bf*)(p + khalf * 2);
            u.h[1] = *(const v8bf*)(p + (khalf + 16) * 2);
            afrag[i] = u.v;
        }
#pragma unroll
        for (int j = 0; j < 2; ++j) {       // B: 32x16 frags from [N][K] layout
            const char* p = Bs + (waveN * 32 + j * 16 + mrow) * LDT;
            union { v16bf v; v8bf h[2]; } u;
            u.h[0] = *(const v8bf*)(p + khalf * 2);
            u.h[1] = *(const v8bf*)(p + (khalf + 16) * 2);
            bfrag[j] = u.v;
        }
#pragma unroll
        for (int i = 0; i < 4; ++i)
#pragma unroll
            for (int j = 0; j < 2; ++j)
                acc[i][j] = __builtin_amdgcn_wmma_f32_16x16x32_bf16(
                    false, afrag[i], false, bfrag[j],
                    (short)0, acc[i][j], false, false);

        async_wait_all();
        __syncthreads();
    }

    // Epilogue. C frag layout: VGPR e -> M = e (+8 for lanes 16-31), N = lane&15.
    const int nofs = lane & 15;
    const int msel = (lane >> 4) * 8;
#pragma unroll
    for (int i = 0; i < 4; ++i) {
        const int gm0 = bm + waveM * 64 + i * 16 + msel;
#pragma unroll
        for (int j = 0; j < 2; ++j) {
            const int gn = bn + waveN * 32 + j * 16 + nofs;
            const float bv = bias[gn];
#pragma unroll
            for (int e = 0; e < 8; ++e) {
                const int gm = gm0 + e;
                float x = acc[i][j][e] + bv;
                if (EPI == 0) {
                    float g = 0.5f * x *
                        (1.0f + tanhf(0.7978845608f * (x + 0.044715f * x * x * x)));
                    Cb[(size_t)gm * N + gn] = (__bf16)g;
                } else {
                    Cf[(size_t)gm * N + gn] = x + resid[(size_t)gm * N + gn];
                }
            }
        }
    }
}

// ---------------------------------------------------------------------------
extern "C" void kernel_launch(void* const* d_in, const int* in_sizes, int n_in,
                              void* d_out, int out_size, void* d_ws, size_t ws_size,
                              hipStream_t stream)
{
    (void)in_sizes; (void)n_in; (void)out_size; (void)ws_size;
    const float* input    = (const float*)d_in[0];
    const float* residual = (const float*)d_in[1];
    /* d_in[2] residual_norm unused by reference */
    const float* bias     = (const float*)d_in[3];
    const float* attn_nw  = (const float*)d_in[4];
    const float* attn_nb  = (const float*)d_in[5];
    const float* inter_w  = (const float*)d_in[6];   // [H][I] fp32
    const float* inter_b  = (const float*)d_in[7];
    const float* output_w = (const float*)d_in[8];   // [I][H] fp32
    const float* output_b = (const float*)d_in[9];
    float* out = (float*)d_out;

    char* ws = (char*)d_ws;
    float*  res_ws = (float*) (ws);                        // 128 MB  [M][H] fp32
    __bf16* ln_ws  = (__bf16*)(ws + 134217728ull);         //  64 MB  [M][H] bf16
    __bf16* w1t    = (__bf16*)(ws + 201326592ull);         // 128 MB  [I][H] bf16
    __bf16* w2t    = (__bf16*)(ws + 335544320ull);         // 128 MB  [H][I] bf16
    __bf16* inter  = (__bf16*)(ws + 469762048ull);         // 256 MB  [M][I] bf16

    // 1) fused residual-add + LayerNorm (fp32 res + bf16 ln)
    add_ln_kernel<<<MTOT, 256, 0, stream>>>(
        input, residual, bias, attn_nw, attn_nb, res_ws, ln_ws);

    // 2) weights fp32 [K][N] -> bf16 [N][K] (half the streamed bytes; L2-resident)
    transpose_bf16_kernel<<<dim3(INTER / 32, HIDDEN / 32), dim3(32, 8), 0, stream>>>(
        inter_w, w1t, HIDDEN, INTER);
    transpose_bf16_kernel<<<dim3(HIDDEN / 32, INTER / 32), dim3(32, 8), 0, stream>>>(
        output_w, w2t, INTER, HIDDEN);

    // 3) GEMM1 + bias + GELU -> bf16 intermediate
    gemm_wmma_kernel<0><<<dim3(INTER / 128, MTOT / 128), 256, 0, stream>>>(
        ln_ws, w1t, inter_b, nullptr, nullptr, inter, MTOT, INTER, HIDDEN);

    // 4) GEMM2 + output_b + res -> fp32 out
    gemm_wmma_kernel<1><<<dim3(HIDDEN / 128, MTOT / 128), 256, 0, stream>>>(
        inter, w2t, output_b, res_ws, out, nullptr, MTOT, HIDDEN, INTER);
}
